// LHATransformerAttention_51479478010640
// MI455X (gfx1250) — matrix operand order repaired and checked by reading, not compiled, not run on hardware
//
#include <hip/hip_runtime.h>
#include <hip/hip_bf16.h>

// ---------------------------------------------------------------------------
// LHA clustered attention for MI455X (gfx1250, wave32, WMMA).
// Matrix work -> v_wmma_f32_16x16x32_f16; tile staging -> async global->LDS
// (GLOBAL_LOAD_ASYNC_TO_LDS_B128, ASYNCcnt path) with vector-copy fallback.
// ---------------------------------------------------------------------------

typedef _Float16 v16h __attribute__((ext_vector_type(16)));
typedef _Float16 v8h  __attribute__((ext_vector_type(8)));
typedef float    v8f  __attribute__((ext_vector_type(8)));

#define Bn  4
#define Ln  4096
#define DMn 1024
#define Hn  16
#define DHn 64
#define Kb  8
#define Cn  512   // Ln / Kb

#if defined(__has_builtin)
#  if __has_builtin(__builtin_amdgcn_global_load_async_to_lds_b128)
#    define HAVE_ASYNC_LDS 1
#  endif
#  if __has_builtin(__builtin_amdgcn_s_wait_asynccnt)
#    define HAVE_WAIT_ASYNC 1
#  endif
#endif

#ifdef HAVE_ASYNC_LDS
// Builtin prototype (from hipcc diagnostic): param0 = int4-vector addrspace(1)*
// (printed "__device__"), param1 = int4-vector addrspace(3)* ("__shared__").
typedef int gvec4i __attribute__((vector_size(16)));
typedef __attribute__((address_space(1))) gvec4i* async_gptr_t;
typedef __attribute__((address_space(3))) gvec4i* async_lptr_t;
#endif

// 16B global -> LDS copy; async (GLOBAL_LOAD_ASYNC_TO_LDS_B128) when possible.
__device__ __forceinline__ void copy16B_to_lds(const void* g, void* l) {
#ifdef HAVE_ASYNC_LDS
    __builtin_amdgcn_global_load_async_to_lds_b128((async_gptr_t)g, (async_lptr_t)l,
                                                   0, 0);
#else
    *(uint4*)l = *(const uint4*)g;
#endif
}

__device__ __forceinline__ void wait_async_copies() {
#ifdef HAVE_WAIT_ASYNC
    __builtin_amdgcn_s_wait_asynccnt(0);
#endif
}

// ---------------- fragment loaders (contiguous k order, same in A and B) ----------
// A 16x32 f16: lane m = lane%16; halfsel = lane/16.
//   elements i: k = kbase + 16*halfsel + i (contiguous, matches frag_b)
__device__ __forceinline__ v16h frag_a(const _Float16* rowp, int kbase, int halfsel) {
    const v8h lo = *(const v8h*)(rowp + kbase + 16 * halfsel);
    const v8h hi = *(const v8h*)(rowp + kbase + 16 * halfsel + 8);
    v16h r;
#pragma unroll
    for (int i = 0; i < 8; ++i) { r[i] = lo[i]; r[i + 8] = hi[i]; }
    return r;
}

// B 32x16 f16: lane n = lane%16; elements i: k = kbase + 16*halfsel + i (contiguous).
// rowp points at the start of the transposed-LDS row for column n (k contiguous).
__device__ __forceinline__ v16h frag_b(const _Float16* rowp, int kbase, int halfsel) {
    const v8h lo = *(const v8h*)(rowp + kbase + 16 * halfsel);
    const v8h hi = *(const v8h*)(rowp + kbase + 16 * halfsel + 8);
    v16h r;
#pragma unroll
    for (int i = 0; i < 8; ++i) { r[i] = lo[i]; r[i + 8] = hi[i]; }
    return r;
}

__device__ __forceinline__ v8f wmma32(v16h a, v16h b, v8f c) {
    return __builtin_amdgcn_wmma_f32_16x16x32_f16(false, a, false, b, (short)0, c,
                                                  false, false);
}

// ---------------- utility kernels -------------------------------------------
__global__ void cvt_f32_to_f16(const float* __restrict__ s, _Float16* __restrict__ d,
                               int n) {
    int i = blockIdx.x * blockDim.x + threadIdx.x;
    int stride = gridDim.x * blockDim.x;
    for (; i < n; i += stride) d[i] = (_Float16)s[i];
}

__global__ void zero_f32(float* __restrict__ p, int n) {
    int i = blockIdx.x * blockDim.x + threadIdx.x;
    if (i < n) p[i] = 0.0f;
}

// ---------------- WMMA GEMM: C[M,N] = A[M,K]*B[K,N] + bias[N] ---------------
// A,B f16 row-major. Output either f16 (C16) or f32 (C32). 64x64 tile per WG.
__global__ __launch_bounds__(256) void gemm_wmma(const _Float16* __restrict__ A,
                                                 const _Float16* __restrict__ Bw,
                                                 const float* __restrict__ bias,
                                                 _Float16* __restrict__ C16,
                                                 float* __restrict__ C32,
                                                 int M, int N, int Kd) {
    __shared__ _Float16 As[64][64];   // [m][k]
    __shared__ _Float16 Bt[64][64];   // transposed: [n][k]

    const int nb = blockIdx.x, mb = blockIdx.y;
    const int tid = threadIdx.x;
    const int lane = tid & 31, wave = tid >> 5;
    const int halfsel = lane >> 4, ln16 = lane & 15;

    v8f acc[2];
#pragma unroll
    for (int t = 0; t < 2; ++t)
#pragma unroll
        for (int j = 0; j < 8; ++j) acc[t][j] = 0.0f;

    const int r = tid >> 2, part = tid & 3;   // 64 rows x 4 loaders
    const int nkb = Kd / 64;
    for (int kb = 0; kb < nkb; ++kb) {
        // stage A tile (async global->LDS, 32B per thread)
        {
            const _Float16* srcA = A + (size_t)(mb * 64 + r) * Kd + kb * 64 + part * 16;
            copy16B_to_lds(srcA, &As[r][part * 16]);
            copy16B_to_lds(srcA + 8, &As[r][part * 16 + 8]);
        }
        // stage B tile transposed ([k][n] global -> [n][k] LDS)
        {
            const _Float16* src = Bw + (size_t)(kb * 64 + r) * N + nb * 64;
#pragma unroll
            for (int c = part * 16; c < part * 16 + 16; ++c) Bt[c][r] = src[c];
        }
        // prefetch next k-block while this one is consumed
        if (kb + 1 < nkb) {
            __builtin_prefetch(A + (size_t)(mb * 64 + r) * Kd + (kb + 1) * 64 + part * 16,
                               0, 1);
            __builtin_prefetch(Bw + (size_t)((kb + 1) * 64 + r) * N + nb * 64 + part * 16,
                               0, 1);
        }
        wait_async_copies();
        __syncthreads();
#pragma unroll
        for (int t = 0; t < 2; ++t) {
            const int tile = wave * 2 + t;
            const int mt = tile >> 2, nt = tile & 3;
            v8f c = acc[t];
#pragma unroll
            for (int kk = 0; kk < 2; ++kk) {
                v16h a = frag_a(&As[mt * 16 + ln16][0], kk * 32, halfsel);
                v16h b = frag_b(&Bt[nt * 16 + ln16][0], kk * 32, halfsel);
                c = wmma32(a, b, c);
            }
            acc[t] = c;
        }
        __syncthreads();
    }

    // epilogue: C/D layout m = j + 8*halfsel, n = lane%16
#pragma unroll
    for (int t = 0; t < 2; ++t) {
        const int tile = wave * 2 + t;
        const int mt = tile >> 2, nt = tile & 3;
#pragma unroll
        for (int j = 0; j < 8; ++j) {
            const int row = mb * 64 + mt * 16 + j + 8 * halfsel;
            const int col = nb * 64 + nt * 16 + ln16;
            const float v = acc[t][j] + bias[col];
            if (C32) C32[(size_t)row * N + col] = v;
            else     C16[(size_t)row * N + col] = (_Float16)v;
        }
    }
}

// ---------------- bucket scores, argmax, aux-loss accumulation --------------
// accs layout: [0..511]=pq_sum, [512..1023]=fq_cnt, [1024..1535]=pk_sum, [1536..2047]=fk_cnt
__global__ void scores_kernel(const _Float16* __restrict__ q16,
                              const _Float16* __restrict__ k16,
                              const float* __restrict__ w_shared,
                              const float* __restrict__ w_qscore,
                              const float* __restrict__ w_kscore,
                              int* __restrict__ bqid, int* __restrict__ bkid,
                              float* __restrict__ accs) {
    const int idx = blockIdx.x * blockDim.x + threadIdx.x;   // over B*L*H
    if (idx >= Bn * Ln * Hn) return;
    const int h = idx % Hn;
    const int l = (idx / Hn) % Ln;
    const int b = idx / (Hn * Ln);

    const _Float16* qr = q16 + (((size_t)b * Ln + l) * Hn + h) * DHn;
    const _Float16* kr = k16 + (((size_t)b * Ln + l) * Hn + h) * DHn;

    float sq[Kb], sk[Kb];
#pragma unroll
    for (int k = 0; k < Kb; ++k) { sq[k] = 0.f; sk[k] = 0.f; }

    for (int d = 0; d < DHn; ++d) {
        const float qv = (float)qr[d], kv = (float)kr[d];
        const float* ws  = w_shared + ((size_t)h * DHn + d) * Kb;
        const float* wqs = w_qscore + ((size_t)h * DHn + d) * Kb;
        const float* wks = w_kscore + ((size_t)h * DHn + d) * Kb;
#pragma unroll
        for (int k = 0; k < Kb; ++k) {
            sq[k] += qv * (0.9f * ws[k] + 0.1f * wqs[k]);
            sk[k] += kv * (0.9f * ws[k] + 0.1f * wks[k]);
        }
    }
    // argmax (first max, matching jnp.argmax)
    int aq = 0, ak = 0;
    float mq = sq[0], mk = sk[0];
#pragma unroll
    for (int k = 1; k < Kb; ++k) {
        if (sq[k] > mq) { mq = sq[k]; aq = k; }
        if (sk[k] > mk) { mk = sk[k]; ak = k; }
    }
    bqid[((size_t)b * Hn + h) * Ln + l] = aq;
    bkid[((size_t)b * Hn + h) * Ln + l] = ak;

    float eq = 0.f, ek = 0.f, pq[Kb], pk[Kb];
#pragma unroll
    for (int k = 0; k < Kb; ++k) {
        pq[k] = __expf(sq[k] - mq); eq += pq[k];
        pk[k] = __expf(sk[k] - mk); ek += pk[k];
    }
    const int bhk = (b * Hn + h) * Kb;
#pragma unroll
    for (int k = 0; k < Kb; ++k) {
        atomicAdd(&accs[bhk + k],        pq[k] / eq);
        atomicAdd(&accs[1024 + bhk + k], pk[k] / ek);
    }
    atomicAdd(&accs[512 + bhk + aq], 1.0f);
    atomicAdd(&accs[1536 + bhk + ak], 1.0f);
}

// ---------------- stable counting sort per (b,h) ----------------------------
__global__ __launch_bounds__(256) void bucket_sort_kernel(const int* __restrict__ ids,
                                                          int* __restrict__ sortOut) {
    const int bh = blockIdx.x;
    const int* id = ids + (size_t)bh * Ln;
    int* out = sortOut + (size_t)bh * Ln;
    __shared__ int hist[Kb];
    __shared__ int offs[Kb];
    const int tid = threadIdx.x;
    if (tid < Kb) hist[tid] = 0;
    __syncthreads();
    for (int i = tid; i < Ln; i += 256) atomicAdd(&hist[id[i]], 1);
    __syncthreads();
    if (tid == 0) {
        int s = 0;
#pragma unroll
        for (int k = 0; k < Kb; ++k) { offs[k] = s; s += hist[k]; }
        int cur[Kb];
#pragma unroll
        for (int k = 0; k < Kb; ++k) cur[k] = offs[k];
        for (int i = 0; i < Ln; ++i) {   // stable placement
            const int bk = id[i];
            out[cur[bk]++] = i;
        }
    }
}

// ---------------- chunk attention (flash-style, WMMA) -----------------------
// grid.x = B*H*K * 8; each WG: 64 sorted query rows of one 512-row chunk.
__global__ __launch_bounds__(256) void attn_kernel(const _Float16* __restrict__ q16,
                                                   const _Float16* __restrict__ k16,
                                                   const _Float16* __restrict__ v16,
                                                   const int* __restrict__ sort_q,
                                                   const int* __restrict__ sort_k,
                                                   _Float16* __restrict__ o16) {
    const int wg = blockIdx.x;
    const int chunk = wg >> 3;            // 0..511
    const int rg = wg & 7;
    const int b = chunk / (Hn * Kb);
    const int h = (chunk / Kb) % Hn;
    const int n = chunk % Kb;
    const int sbase = (b * Hn + h) * Ln + n * Cn;
    const int row0 = rg * 64;

    const int tid = threadIdx.x;
    const int lane = tid & 31, wave = tid >> 5;
    const int halfsel = lane >> 4, ln16 = lane & 15;

    __shared__ _Float16 Qs[64][64];       // [qrow][d]
    __shared__ _Float16 Ks[64][64];       // [krow][d]
    __shared__ _Float16 Vt[64][64];       // transposed: [d][krow]
    __shared__ float    Ss[64][64];       // logits block
    __shared__ _Float16 Ps[64][64];       // probabilities block (f16)
    __shared__ float rowMax[64], rowSum[64], rowAlpha[64];
    __shared__ int qtok[64];

    const int r = tid >> 2, part = tid & 3;
    // gather Q rows via sort_q (async global->LDS)
    {
        const int tq = sort_q[sbase + row0 + r];
        if (part == 0) qtok[r] = tq;
        const _Float16* src = q16 + (((size_t)b * Ln + tq) * Hn + h) * DHn + part * 16;
        copy16B_to_lds(src, &Qs[r][part * 16]);
        copy16B_to_lds(src + 8, &Qs[r][part * 16 + 8]);
    }
    if (tid < 64) { rowMax[tid] = -3.0e38f; rowSum[tid] = 0.0f; }

    v8f oacc[2];
#pragma unroll
    for (int t = 0; t < 2; ++t)
#pragma unroll
        for (int j = 0; j < 8; ++j) oacc[t][j] = 0.0f;
    wait_async_copies();
    __syncthreads();

    const float scale = 0.125f;   // 1/sqrt(64)
    for (int kb = 0; kb < 8; ++kb) {
        // gather K block (async) + V block (transposed, scalar) via sort_k
        {
            const int tk = sort_k[sbase + kb * 64 + r];
            const _Float16* srck =
                k16 + (((size_t)b * Ln + tk) * Hn + h) * DHn + part * 16;
            copy16B_to_lds(srck, &Ks[r][part * 16]);
            copy16B_to_lds(srck + 8, &Ks[r][part * 16 + 8]);
            const _Float16* srcv = v16 + (((size_t)b * Ln + tk) * Hn + h) * DHn;
#pragma unroll
            for (int d = part * 16; d < part * 16 + 16; ++d) Vt[d][r] = srcv[d];
        }
        wait_async_copies();
        __syncthreads();

        // S = Q * K^T  (16 tiles of 16x16, 2 per wave)
#pragma unroll
        for (int t = 0; t < 2; ++t) {
            const int tile = wave * 2 + t;
            const int mt = tile >> 2, nt = tile & 3;
            v8f c;
#pragma unroll
            for (int j = 0; j < 8; ++j) c[j] = 0.0f;
#pragma unroll
            for (int kk = 0; kk < 2; ++kk) {
                v16h a = frag_a(&Qs[mt * 16 + ln16][0], kk * 32, halfsel);
                v16h bb = frag_b(&Ks[nt * 16 + ln16][0], kk * 32, halfsel); // B[k][n]=K[n][k]
                c = wmma32(a, bb, c);
            }
#pragma unroll
            for (int j = 0; j < 8; ++j)
                Ss[mt * 16 + j + 8 * halfsel][nt * 16 + ln16] = c[j] * scale;
        }
        __syncthreads();

        // online softmax row update
        if (tid < 64) {
            const float m0 = rowMax[tid];
            float bm = m0;
            for (int c = 0; c < 64; ++c) bm = fmaxf(bm, Ss[tid][c]);
            const float alpha = __expf(m0 - bm);
            float s = 0.f;
            for (int c = 0; c < 64; ++c) {
                const float p = __expf(Ss[tid][c] - bm);
                Ps[tid][c] = (_Float16)p;
                s += p;
            }
            rowSum[tid] = rowSum[tid] * alpha + s;
            rowMax[tid] = bm;
            rowAlpha[tid] = alpha;
        }
        __syncthreads();

        // O = alpha*O + P * V
#pragma unroll
        for (int t = 0; t < 2; ++t) {
            const int tile = wave * 2 + t;
            const int mt = tile >> 2, nt = tile & 3;
            v8f c = oacc[t];
#pragma unroll
            for (int j = 0; j < 8; ++j)
                c[j] *= rowAlpha[mt * 16 + j + 8 * halfsel];
#pragma unroll
            for (int kk = 0; kk < 2; ++kk) {
                v16h a = frag_a(&Ps[mt * 16 + ln16][0], kk * 32, halfsel);
                v16h bb = frag_b(&Vt[nt * 16 + ln16][0], kk * 32, halfsel); // B[k][n]=Vt[n][k]
                c = wmma32(a, bb, c);
            }
            oacc[t] = c;
        }
        __syncthreads();
    }

    // normalize + unsort-scatter (token = sort_q[pos] reverses reference inv_q gather)
#pragma unroll
    for (int t = 0; t < 2; ++t) {
        const int tile = wave * 2 + t;
        const int mt = tile >> 2, nt = tile & 3;
#pragma unroll
        for (int j = 0; j < 8; ++j) {
            const int rl = mt * 16 + j + 8 * halfsel;
            const float inv = 1.0f / rowSum[rl];
            const int tq = qtok[rl];
            o16[(((size_t)b * Ln + tq) * Hn + h) * DHn + nt * 16 + ln16] =
                (_Float16)(oacc[t][j] * inv);
        }
    }
}

// ---------------- aux routing loss finalize ---------------------------------
__global__ void loss_kernel(const float* __restrict__ accs, float* __restrict__ out) {
    __shared__ float rq[64], rk[64];
    const int t = threadIdx.x;   // 64 threads = (b,h)
    const float invL = 1.0f / (float)Ln;
    float sqv = 0.f, skv = 0.f;
#pragma unroll
    for (int k = 0; k < Kb; ++k) {
        sqv += (accs[512 + t * Kb + k] * invL) * (accs[t * Kb + k] * invL);
        skv += (accs[1536 + t * Kb + k] * invL) * (accs[1024 + t * Kb + k] * invL);
    }
    rq[t] = sqv; rk[t] = skv;
    __syncthreads();
    if (t == 0) {
        float aq = 0.f, ak = 0.f;
        for (int i = 0; i < 64; ++i) { aq += rq[i]; ak += rk[i]; }
        const float lq = (float)Kb * (aq / 64.0f);
        const float lk = (float)Kb * (ak / 64.0f);
        out[0] = 0.5f * (lq + lk);
    }
}

// ---------------------------------------------------------------------------
extern "C" void kernel_launch(void* const* d_in, const int* in_sizes, int n_in,
                              void* d_out, int out_size, void* d_ws, size_t ws_size,
                              hipStream_t stream) {
    const float* inputs_q = (const float*)d_in[0];
    const float* wq = (const float*)d_in[1];
    const float* bq = (const float*)d_in[2];
    const float* wk = (const float*)d_in[3];
    const float* bk = (const float*)d_in[4];
    const float* wv = (const float*)d_in[5];
    const float* bv = (const float*)d_in[6];
    const float* w_shared = (const float*)d_in[7];
    const float* w_qscore = (const float*)d_in[8];
    const float* w_kscore = (const float*)d_in[9];
    const float* wo = (const float*)d_in[10];
    const float* bo = (const float*)d_in[11];
    float* out = (float*)d_out;   // [B*L*DM] + [1] extra_loss

    char* ws = (char*)d_ws;
    size_t off = 0;
    auto alloc = [&](size_t bytes) -> void* {
        void* p = ws + off;
        off += (bytes + 255) & ~(size_t)255;
        return p;
    };
    const size_t M = (size_t)Bn * Ln;           // 16384
    _Float16* x16  = (_Float16*)alloc(M * DMn * 2);
    _Float16* wq16 = (_Float16*)alloc((size_t)DMn * Hn * DHn * 2);
    _Float16* wk16 = (_Float16*)alloc((size_t)DMn * Hn * DHn * 2);
    _Float16* wv16 = (_Float16*)alloc((size_t)DMn * Hn * DHn * 2);
    _Float16* wo16 = (_Float16*)alloc((size_t)Hn * DHn * DMn * 2);
    _Float16* q16  = (_Float16*)alloc(M * Hn * DHn * 2);
    _Float16* k16  = (_Float16*)alloc(M * Hn * DHn * 2);
    _Float16* v16  = (_Float16*)alloc(M * Hn * DHn * 2);
    _Float16* o16  = (_Float16*)alloc(M * Hn * DHn * 2);
    int* bqid  = (int*)alloc((size_t)Bn * Hn * Ln * 4);
    int* bkid  = (int*)alloc((size_t)Bn * Hn * Ln * 4);
    int* sortq = (int*)alloc((size_t)Bn * Hn * Ln * 4);
    int* sortk = (int*)alloc((size_t)Bn * Hn * Ln * 4);
    float* accs = (float*)alloc(2048 * 4);

    // 0) zero aux-loss accumulators (every call)
    zero_f32<<<8, 256, 0, stream>>>(accs, 2048);

    // 1) f32 -> f16 conversions
    cvt_f32_to_f16<<<2048, 256, 0, stream>>>(inputs_q, x16, (int)(M * DMn));
    cvt_f32_to_f16<<<512, 256, 0, stream>>>(wq, wq16, DMn * Hn * DHn);
    cvt_f32_to_f16<<<512, 256, 0, stream>>>(wk, wk16, DMn * Hn * DHn);
    cvt_f32_to_f16<<<512, 256, 0, stream>>>(wv, wv16, DMn * Hn * DHn);
    cvt_f32_to_f16<<<512, 256, 0, stream>>>(wo, wo16, Hn * DHn * DMn);

    // 2) QKV projections (WMMA GEMM, f16 output)
    dim3 ggrid((Hn * DHn) / 64, (unsigned)(M / 64));   // (16, 256)
    gemm_wmma<<<ggrid, 256, 0, stream>>>(x16, wq16, bq, q16, nullptr,
                                         (int)M, Hn * DHn, DMn);
    gemm_wmma<<<ggrid, 256, 0, stream>>>(x16, wk16, bk, k16, nullptr,
                                         (int)M, Hn * DHn, DMn);
    gemm_wmma<<<ggrid, 256, 0, stream>>>(x16, wv16, bv, v16, nullptr,
                                         (int)M, Hn * DHn, DMn);

    // 3) bucket scores + argmax + loss accumulation
    scores_kernel<<<(Bn * Ln * Hn) / 256, 256, 0, stream>>>(
        q16, k16, w_shared, w_qscore, w_kscore, bqid, bkid, accs);

    // 4) stable counting sort per (b,h)
    bucket_sort_kernel<<<Bn * Hn, 256, 0, stream>>>(bqid, sortq);
    bucket_sort_kernel<<<Bn * Hn, 256, 0, stream>>>(bkid, sortk);

    // 5) chunked flash attention (WMMA)
    attn_kernel<<<Bn * Hn * Kb * 8, 256, 0, stream>>>(q16, k16, v16, sortq, sortk, o16);

    // 6) output projection (WMMA GEMM, f32 output directly into d_out)
    gemm_wmma<<<ggrid, 256, 0, stream>>>(o16, wo16, bo, nullptr, out,
                                         (int)M, DMn, Hn * DHn);

    // 7) aux loss
    loss_kernel<<<1, 64, 0, stream>>>(accs, out + (size_t)Bn * Ln * DMn);
}
